// FCGF_MLP3_89575837925656
// MI455X (gfx1250) — matrix operand order repaired and checked by reading, not compiled, not run on hardware
//
#include <hip/hip_runtime.h>
#include <hip/hip_bf16.h>

typedef __attribute__((ext_vector_type(16))) __bf16 v16bf;
typedef __attribute__((ext_vector_type(8)))  __bf16 v8bf;
typedef __attribute__((ext_vector_type(8)))  float  v8f;

#define NPTS 524288
#define NT   (NPTS / 16)
#define NSEG 64

// workspace byte offsets
#define OFF_W1BF 0          //  64*32 bf16  = 4096 B
#define OFF_W2BF 4096       // 256*64 bf16  = 32768 B
#define OFF_SUMS 36864      // sum1[64], sq1[64], sum2[256], sq2[256] = 640 f
#define OFF_A1   39424
#define OFF_S1   39680
#define OFF_A2   39936
#define OFF_S2   40960
#define OFF_CUM  41984      // int cum[65]
#define OFF_SEG  42496      // float segsum[64*256]

static __device__ __forceinline__ void atomicAddF(float* p, float v) {
  unsafeAtomicAdd(p, v);   // native global_atomic_add_f32
}

static __device__ __forceinline__ v8f wmma_bf16(v16bf a, v16bf b, v8f c) {
  // D = A(16x32 bf16) x B(32x16 bf16) + C(16x16 f32)
  return __builtin_amdgcn_wmma_f32_16x16x32_bf16(false, a, false, b, (short)0, c,
                                                 false, false);
}

// Build A-operand (16x32 bf16) for a 16-row tile of x [N,32] f32, row-major.
// A layout: lane l<16: M=l, K={0..15}; lane>=16: M=l-16, K={16..31}.
static __device__ __forceinline__ v16bf load_a1(const float* __restrict__ x,
                                                int rowBase, int lane) {
  const float* p = x + (rowBase + (lane & 15)) * 32 + ((lane >> 4) << 4);
  float4 c0 = *(const float4*)(p + 0);
  float4 c1 = *(const float4*)(p + 4);
  float4 c2 = *(const float4*)(p + 8);
  float4 c3 = *(const float4*)(p + 12);
  v16bf a;
  a[0]=(__bf16)c0.x;  a[1]=(__bf16)c0.y;  a[2]=(__bf16)c0.z;  a[3]=(__bf16)c0.w;
  a[4]=(__bf16)c1.x;  a[5]=(__bf16)c1.y;  a[6]=(__bf16)c1.z;  a[7]=(__bf16)c1.w;
  a[8]=(__bf16)c2.x;  a[9]=(__bf16)c2.y;  a[10]=(__bf16)c2.z; a[11]=(__bf16)c2.w;
  a[12]=(__bf16)c3.x; a[13]=(__bf16)c3.y; a[14]=(__bf16)c3.z; a[15]=(__bf16)c3.w;
  return a;
}

// ---------------------------------------------------------------- prep
__global__ void __launch_bounds__(256)
k_prep(const float* __restrict__ W1, const float* __restrict__ W2,
       const int* __restrict__ length, __bf16* __restrict__ W1bf,
       __bf16* __restrict__ W2bf, int* __restrict__ cum,
       float* __restrict__ sums, float* __restrict__ segsum) {
  const int tid = threadIdx.x;
  for (int i = tid; i < 64 * 32; i += blockDim.x)  W1bf[i] = (__bf16)W1[i];
  for (int i = tid; i < 256 * 64; i += blockDim.x) W2bf[i] = (__bf16)W2[i];
  for (int i = tid; i < 640; i += blockDim.x)      sums[i] = 0.f;
  for (int i = tid; i < 64 * 256; i += blockDim.x) segsum[i] = 0.f;
  if (tid == 0) {
    int acc = 0; cum[0] = 0;
    for (int i = 0; i < NSEG; ++i) { acc += length[i]; cum[i + 1] = acc; }
  }
}

// ---------------------------------------------------------------- pass 1: stats of h1
__global__ void __launch_bounds__(256)
k_stats1(const float* __restrict__ x, const __bf16* __restrict__ W1bf,
         float* __restrict__ sum1, float* __restrict__ sq1) {
  const int lane = threadIdx.x & 31;
  const int gw = (blockIdx.x * blockDim.x + threadIdx.x) >> 5;
  const int nw = (gridDim.x * blockDim.x) >> 5;
  const int nn = lane & 15, kb2 = (lane >> 4) << 4;
  v16bf b1[4];
#pragma unroll
  for (int t = 0; t < 4; ++t)
    b1[t] = *(const v16bf*)(W1bf + (16 * t + nn) * 32 + kb2);

  float s4[4] = {0.f, 0.f, 0.f, 0.f}, q4[4] = {0.f, 0.f, 0.f, 0.f};
  for (int tile = gw; tile < NT; tile += nw) {
    v16bf a = load_a1(x, tile * 16, lane);
#pragma unroll
    for (int t = 0; t < 4; ++t) {
      v8f d = {};
      d = wmma_bf16(a, b1[t], d);
      float ps = 0.f, pq = 0.f;
#pragma unroll
      for (int j = 0; j < 8; ++j) { float v = d[j]; ps += v; pq += v * v; }
      s4[t] += ps; q4[t] += pq;
    }
  }
#pragma unroll
  for (int t = 0; t < 4; ++t) {
    float ss = s4[t] + __shfl_xor(s4[t], 16);
    float qq = q4[t] + __shfl_xor(q4[t], 16);
    if (lane < 16) {
      atomicAddF(&sum1[16 * t + lane], ss);
      atomicAddF(&sq1[16 * t + lane], qq);
    }
  }
}

// ---------------------------------------------------------------- finalize BN coeffs
__global__ void k_final(const float* __restrict__ sums, const float* __restrict__ sqs,
                        const float* __restrict__ g, const float* __restrict__ beta,
                        float* __restrict__ a, float* __restrict__ s, int C) {
  const int c = threadIdx.x;
  if (c < C) {
    const float inv = 1.f / (float)NPTS;
    float m = sums[c] * inv;
    float var = fmaxf(sqs[c] * inv - m * m, 0.f);
    float av = g[c] * rsqrtf(var + 1e-5f);
    a[c] = av;
    s[c] = beta[c] - av * m;   // biases cancel inside BN
  }
}

// ---------------------------------------------------------------- pass 2: stats of h2
__global__ void __launch_bounds__(256)
k_stats2(const float* __restrict__ x, const __bf16* __restrict__ W1bf,
         const __bf16* __restrict__ W2bf, const float* __restrict__ a1v,
         const float* __restrict__ s1v, float* __restrict__ sum2,
         float* __restrict__ sq2) {
  __shared__ __bf16 ldsT[8 * 16 * 64];   // per-wave 16x64 bf16 transpose buffer
  const int lane = threadIdx.x & 31;
  const int wib = threadIdx.x >> 5;
  const int gw = (blockIdx.x * blockDim.x + threadIdx.x) >> 5;
  const int half = gw & 1;               // channel half of GEMM2
  const int slot = gw >> 1;
  const int nslot = ((gridDim.x * blockDim.x) >> 5) >> 1;
  const int nn = lane & 15, kb2 = (lane >> 4) << 4, rb = (lane >> 4) << 3;
  __bf16* myT = &ldsT[wib * (16 * 64)];

  v16bf b1[4]; float a1c[4], s1c[4];
#pragma unroll
  for (int t = 0; t < 4; ++t) {
    b1[t] = *(const v16bf*)(W1bf + (16 * t + nn) * 32 + kb2);
    a1c[t] = a1v[16 * t + nn];
    s1c[t] = s1v[16 * t + nn];
  }
  float s8[8] = {0}, q8[8] = {0};

  for (int tile = slot; tile < NT; tile += nslot) {
    v16bf a = load_a1(x, tile * 16, lane);
#pragma unroll
    for (int t = 0; t < 4; ++t) {          // GEMM1 + bnrelu1 + transpose
      v8f d = {};
      d = wmma_bf16(a, b1[t], d);
      const int c = 16 * t + nn;
#pragma unroll
      for (int j = 0; j < 8; ++j) {
        float h = fmaxf(a1c[t] * d[j] + s1c[t], 0.f);
        myT[(rb + j) * 64 + c] = (__bf16)h;  // row-major [16][64]
      }
    }
    v16bf a2[2];
#pragma unroll
    for (int st = 0; st < 2; ++st) {       // A operands for GEMM2 (K=64 -> 2 steps)
      const __bf16* pa = myT + nn * 64 + st * 32 + kb2;
      v8bf lo = *(const v8bf*)(pa);
      v8bf hi = *(const v8bf*)(pa + 8);
      a2[st] = __builtin_shufflevector(lo, hi, 0,1,2,3,4,5,6,7,8,9,10,11,12,13,14,15);
    }
#pragma unroll
    for (int tt = 0; tt < 8; ++tt) {       // 8 channel tiles of this half
      const int n2 = 16 * (half * 8 + tt) + nn;
      v16bf bA = *(const v16bf*)(W2bf + n2 * 64 + kb2);
      v16bf bB = *(const v16bf*)(W2bf + n2 * 64 + 32 + kb2);
      v8f d = {};
      d = wmma_bf16(a2[0], bA, d);
      d = wmma_bf16(a2[1], bB, d);
      float ps = 0.f, pq = 0.f;
#pragma unroll
      for (int j = 0; j < 8; ++j) { float v = d[j]; ps += v; pq += v * v; }
      s8[tt] += ps; q8[tt] += pq;
    }
  }
#pragma unroll
  for (int tt = 0; tt < 8; ++tt) {
    float ss = s8[tt] + __shfl_xor(s8[tt], 16);
    float qq = q8[tt] + __shfl_xor(q8[tt], 16);
    if (lane < 16) {
      atomicAddF(&sum2[16 * (half * 8 + tt) + lane], ss);
      atomicAddF(&sq2[16 * (half * 8 + tt) + lane], qq);
    }
  }
}

// ---------------------------------------------------------------- pass 3: full fwd + segment pool
__global__ void __launch_bounds__(256)
k_main(const float* __restrict__ x, const __bf16* __restrict__ W1bf,
       const __bf16* __restrict__ W2bf, const float* __restrict__ a1v,
       const float* __restrict__ s1v, const float* __restrict__ a2v,
       const float* __restrict__ s2v, const int* __restrict__ cum,
       float* __restrict__ segsum) {
  __shared__ __bf16 ldsT[8 * 16 * 64];
  __shared__ int scum[NSEG + 1];
  for (int i = threadIdx.x; i < NSEG + 1; i += blockDim.x) scum[i] = cum[i];
  __syncthreads();

  const int lane = threadIdx.x & 31;
  const int wib = threadIdx.x >> 5;
  const int gw = (blockIdx.x * blockDim.x + threadIdx.x) >> 5;
  const int half = gw & 1;
  const int slot = gw >> 1;
  const int nslot = ((gridDim.x * blockDim.x) >> 5) >> 1;
  const int tps = (NT + nslot - 1) / nslot;
  const int tA = slot * tps;
  const int tB = min(tA + tps, NT);        // contiguous tile range -> rare seg flushes
  const int nn = lane & 15, kb2 = (lane >> 4) << 4, rb = (lane >> 4) << 3;
  __bf16* myT = &ldsT[wib * (16 * 64)];

  v16bf b1[4]; float a1c[4], s1c[4];
#pragma unroll
  for (int t = 0; t < 4; ++t) {
    b1[t] = *(const v16bf*)(W1bf + (16 * t + nn) * 32 + kb2);
    a1c[t] = a1v[16 * t + nn];
    s1c[t] = s1v[16 * t + nn];
  }
  float a2c[8], s2c[8];
#pragma unroll
  for (int tt = 0; tt < 8; ++tt) {
    const int c = 16 * (half * 8 + tt) + nn;
    a2c[tt] = a2v[c]; s2c[tt] = s2v[c];
  }

  float segreg[8] = {0, 0, 0, 0, 0, 0, 0, 0};
  int seg = 0;

  for (int tile = tA; tile < tB; ++tile) {
    const int r0 = tile * 16;
    while (r0 >= scum[seg + 1]) {          // crossed a segment boundary: flush
#pragma unroll
      for (int tt = 0; tt < 8; ++tt) {
        float ls = segreg[tt] + __shfl_xor(segreg[tt], 16);
        if (lane < 16 && ls != 0.f)
          atomicAddF(&segsum[seg * 256 + 16 * (half * 8 + tt) + lane], ls);
        segreg[tt] = 0.f;
      }
      ++seg;
    }
    const bool straddle = (r0 + 15 >= scum[seg + 1]);

    v16bf a = load_a1(x, r0, lane);
#pragma unroll
    for (int t = 0; t < 4; ++t) {
      v8f d = {};
      d = wmma_bf16(a, b1[t], d);
      const int c = 16 * t + nn;
#pragma unroll
      for (int j = 0; j < 8; ++j) {
        float h = fmaxf(a1c[t] * d[j] + s1c[t], 0.f);
        myT[(rb + j) * 64 + c] = (__bf16)h;
      }
    }
    v16bf a2[2];
#pragma unroll
    for (int st = 0; st < 2; ++st) {
      const __bf16* pa = myT + nn * 64 + st * 32 + kb2;
      v8bf lo = *(const v8bf*)(pa);
      v8bf hi = *(const v8bf*)(pa + 8);
      a2[st] = __builtin_shufflevector(lo, hi, 0,1,2,3,4,5,6,7,8,9,10,11,12,13,14,15);
    }
#pragma unroll
    for (int tt = 0; tt < 8; ++tt) {
      const int n2 = 16 * (half * 8 + tt) + nn;
      v16bf bA = *(const v16bf*)(W2bf + n2 * 64 + kb2);
      v16bf bB = *(const v16bf*)(W2bf + n2 * 64 + 32 + kb2);
      v8f d = {};
      d = wmma_bf16(a2[0], bA, d);
      d = wmma_bf16(a2[1], bB, d);
      if (!straddle) {
        float p = 0.f;
#pragma unroll
        for (int j = 0; j < 8; ++j) p += fmaxf(a2c[tt] * d[j] + s2c[tt], 0.f);
        segreg[tt] += p;
      } else {                             // rare boundary tile: per-row segments
        int srow = seg;
#pragma unroll
        for (int j = 0; j < 8; ++j) {
          const int row = r0 + rb + j;
          while (row >= scum[srow + 1]) ++srow;
          float h = fmaxf(a2c[tt] * d[j] + s2c[tt], 0.f);
          atomicAddF(&segsum[srow * 256 + 16 * (half * 8 + tt) + nn], h);
        }
      }
    }
  }
#pragma unroll
  for (int tt = 0; tt < 8; ++tt) {         // final flush
    float ls = segreg[tt] + __shfl_xor(segreg[tt], 16);
    if (lane < 16 && ls != 0.f)
      atomicAddF(&segsum[seg * 256 + 16 * (half * 8 + tt) + lane], ls);
  }
}

// ---------------------------------------------------------------- mean + L2 normalize
__global__ void __launch_bounds__(256)
k_out(const float* __restrict__ segsum, const int* __restrict__ length,
      float* __restrict__ out) {
  __shared__ float red[8];
  const int b = blockIdx.x, c = threadIdx.x;
  float m = segsum[b * 256 + c] / (float)length[b];
  float ss = m * m;
#pragma unroll
  for (int o = 16; o > 0; o >>= 1) ss += __shfl_down(ss, o);
  if ((c & 31) == 0) red[c >> 5] = ss;
  __syncthreads();
  if (c == 0) {
    float t = 0.f;
#pragma unroll
    for (int i = 0; i < 8; ++i) t += red[i];
    red[0] = fmaxf(sqrtf(t), 1e-12f);
  }
  __syncthreads();
  out[b * 256 + c] = m / red[0];
}

// ---------------------------------------------------------------- launcher
extern "C" void kernel_launch(void* const* d_in, const int* in_sizes, int n_in,
                              void* d_out, int out_size, void* d_ws, size_t ws_size,
                              hipStream_t stream) {
  const float* x      = (const float*)d_in[0];
  const int*   length = (const int*)d_in[1];
  const float* W1     = (const float*)d_in[2];
  const float* g1     = (const float*)d_in[4];
  const float* be1    = (const float*)d_in[5];
  const float* W2     = (const float*)d_in[6];
  const float* g2     = (const float*)d_in[8];
  const float* be2    = (const float*)d_in[9];
  float* out = (float*)d_out;

  char* ws = (char*)d_ws;
  __bf16* W1bf = (__bf16*)(ws + OFF_W1BF);
  __bf16* W2bf = (__bf16*)(ws + OFF_W2BF);
  float* sums  = (float*)(ws + OFF_SUMS);
  float* sum1 = sums, *sq1 = sums + 64, *sum2 = sums + 128, *sq2 = sums + 384;
  float* a1 = (float*)(ws + OFF_A1);
  float* s1 = (float*)(ws + OFF_S1);
  float* a2 = (float*)(ws + OFF_A2);
  float* s2 = (float*)(ws + OFF_S2);
  int*   cum = (int*)(ws + OFF_CUM);
  float* segsum = (float*)(ws + OFF_SEG);

  k_prep<<<1, 256, 0, stream>>>(W1, W2, length, W1bf, W2bf, cum, sums, segsum);
  k_stats1<<<256, 256, 0, stream>>>(x, W1bf, sum1, sq1);
  k_final<<<1, 64, 0, stream>>>(sum1, sq1, g1, be1, a1, s1, 64);
  k_stats2<<<256, 256, 0, stream>>>(x, W1bf, W2bf, a1, s1, sum2, sq2);
  k_final<<<1, 256, 0, stream>>>(sum2, sq2, g2, be2, a2, s2, 256);
  k_main<<<256, 256, 0, stream>>>(x, W1bf, W2bf, a1, s1, a2, s2, cum, segsum);
  k_out<<<64, 256, 0, stream>>>(segsum, length, out);
  (void)in_sizes; (void)n_in; (void)out_size; (void)ws_size;
}
